// SelfLayer_44057774522740
// MI455X (gfx1250) — compile-verified, matrix operand order checked by reading, not run on hardware
//
#include <hip/hip_runtime.h>
#include <cmath>

// ---------------------------------------------------------------------------
// MI455X (gfx1250) implementation of the e3nn "SelfLayer" reference.
// All channel-mixing matmuls -> v_wmma_f32_16x16x32_bf16 (bf16 in, f32 acc),
// B operands pre-packed into WMMA fragment layout (streamed from L2, no LDS),
// A tiles staged via double-buffered GLOBAL_LOAD_ASYNC_TO_LDS_B128 (ASYNCcnt)
// where the source is contiguous. CG tensor product / SiLU -> VALU.
// ---------------------------------------------------------------------------

typedef __attribute__((ext_vector_type(16))) __bf16 v16bf;
typedef __attribute__((ext_vector_type(8)))  float  v8f;

#define NODES   16384
#define IN_DIM  1152
#define NM_IN   384
#define NM_TP   1408
#define TP_DIM  4480

// tp path layout: PATHS = [(0,0,0),(0,1,1),(0,2,2),(1,0,1),(1,1,0),(1,1,2),
//                          (1,2,1),(2,0,2),(2,1,1),(2,2,0),(2,2,2)]
__constant__ int c_tp_base[12] = {0,128,512,1152,1536,1664,2304,2688,3328,3712,3840,4480};
__constant__ int c_tp_w[11]    = {1,3,5,3,1,5,3,5,3,1,5};
// linear_p: base offsets of the equal-l tp groups (l=0 has 3 groups, l=1/2 have 4)
__constant__ int c_grp_base[3][4] = {
    {   0, 1536, 3712,    0},   // lo==0: p0,p4,p9
    { 128, 1152, 2304, 3328},   // lo==1: p1,p3,p6,p8
    { 512, 1664, 2688, 3840},   // lo==2: p2,p5,p7,p10
};

__device__ __forceinline__ float silu_f(float v) { return v / (1.f + __expf(-v)); }

// ---------------------------------------------------------------------------
// Pack an f32 row-major weight [K,N] into bf16 WMMA B-fragment layout:
//   dst[((kt*ntiles + nt)*32 + lane)*16 + e] = B[kt*32 + (e&7)+8*(lane>>4)+16*(e>>3),
//                                                nt*16 + (lane&15)]
// A wave's B fragment then becomes one aligned 32-byte contiguous load.
// ---------------------------------------------------------------------------
__global__ void pack_b(const float* __restrict__ src, __bf16* __restrict__ dst,
                       int K, int N) {
    int t = blockIdx.x * blockDim.x + threadIdx.x;
    if (t >= K * N) return;
    int e    = t & 15;
    int lane = (t >> 4) & 31;
    int tile = t >> 9;
    int ntiles = N >> 4;
    int kt = tile / ntiles, nt = tile % ntiles;
    int k = kt * 32 + (e & 7) + 8 * (lane >> 4) + 16 * (e >> 3);
    int n = nt * 16 + (lane & 15);
    dst[t] = (__bf16)src[(size_t)k * N + n];
}

// ---------------------------------------------------------------------------
// Invariants of x: [N,384] bf16  (l=0: value, l>0: L2 norm over 2l+1 comps)
// ---------------------------------------------------------------------------
__global__ void invariants_x(const float* __restrict__ x, __bf16* __restrict__ inv) {
    int t = blockIdx.x * blockDim.x + threadIdx.x;
    if (t >= NODES * NM_IN) return;
    int n = t / NM_IN, c = t % NM_IN;
    int l = c >> 7, u = c & 127;
    const float* xr = x + (size_t)n * IN_DIM;
    float v;
    if (l == 0) v = xr[u];
    else if (l == 1) {
        const float* p = xr + 128 + u * 3;
        v = sqrtf(p[0]*p[0] + p[1]*p[1] + p[2]*p[2]);
    } else {
        const float* p = xr + 512 + u * 5;
        float s = 0.f;
        #pragma unroll
        for (int i = 0; i < 5; ++i) s += p[i]*p[i];
        v = sqrtf(s);
    }
    inv[t] = (__bf16)v;
}

// ---------------------------------------------------------------------------
// Apply both gates to x -> bf16 gated activations (left & right)
// ---------------------------------------------------------------------------
__global__ void apply_gate_lr(const float* __restrict__ x,
                              const float* __restrict__ gl, const float* __restrict__ gr,
                              __bf16* __restrict__ xgl, __bf16* __restrict__ xgr) {
    int t = blockIdx.x * blockDim.x + threadIdx.x;
    if (t >= NODES * IN_DIM) return;
    int n = t / IN_DIM, col = t % IN_DIM;
    int l, u;
    if (col < 128)      { l = 0; u = col; }
    else if (col < 512) { l = 1; u = (col - 128) / 3; }
    else                { l = 2; u = (col - 512) / 5; }
    int gi = n * NM_IN + l * 128 + u;
    float xv = x[t];
    xgl[t] = (__bf16)(xv * gl[gi]);
    xgr[t] = (__bf16)(xv * gr[gi]);
}

// ---------------------------------------------------------------------------
// bf16 WMMA GEMM:  C[M,N] = act(A[M,K] @ B[K,N] * scale + bias)
//   Block = 128 threads / 4 waves; macro-tile M=32 x N=64, K-step 64.
//   Each wave owns one 16-col N tile and TWO 16-row M tiles (two accumulators
//   sharing each B fragment) -> 4 WMMAs between every barrier pair.
//   B is pre-packed fragment-major (see pack_b), read directly from global/L2.
//   ASYNC=1 (requires a_cs==1): A staged with double-buffered
//   global_load_async_to_lds_b128 + s_wait_asynccnt (next tile overlaps MMA).
//   ASYNC=0: synchronous element gather (strided / grouped A views).
//   AMODE 0: A col offset = a_base + c * a_cs
//   AMODE 1: A col offset = grp_base[lsel][c>>7] + (c&127)*a_cs + a_base
//   OUTMODE 0: f32 out + addend   1: bf16 out   2: f32 out       ACT 1: SiLU
// ---------------------------------------------------------------------------
template<int AMODE, int OUTMODE, int ACT, int ASYNC>
__global__ __launch_bounds__(128) void gemm_wmma(
    const __bf16* __restrict__ A, int a_row, int a_base, int a_cs,
    const __bf16* __restrict__ Bpk, int N,
    const float* __restrict__ bias, float scale,
    void* __restrict__ Cp, int c_row, int c_base, int c_cs,
    const float* __restrict__ addend, int K, int lsel)
{
    __shared__ __bf16 Asm[2][32 * 64];       // double-buffered A tile [m][k]

    const int tid    = threadIdx.x;
    const int wave   = tid >> 5;
    const int lane   = tid & 31;
    const int m0     = blockIdx.y * 32;
    const int nt     = blockIdx.x * 4 + wave;   // this wave's 16-col N tile
    const int ntiles = N >> 4;

    v8f acc0 = {}, acc1 = {};

    // 4 WMMAs over one staged 32x64 A tile (two k-slices x two M tiles)
    auto mma_tile = [&](const __bf16* buf, int ktile0) {
        #pragma unroll
        for (int ks = 0; ks < 2; ++ks) {
            const int kt = ktile0 + ks;
            v16bf bfr = *(const v16bf*)(Bpk + (((size_t)kt * ntiles + nt) * 32 + lane) * 16);
            v16bf a0, a1;
            const int ml = lane & 15, h = lane >> 4;
            #pragma unroll
            for (int e = 0; e < 16; ++e) {
                int k = ks * 32 + (e & 7) + 8 * h + 16 * (e >> 3);
                a0[e] = buf[ml * 64 + k];
                a1[e] = buf[(16 + ml) * 64 + k];
            }
            acc0 = __builtin_amdgcn_wmma_f32_16x16x32_bf16(
                       false, a0, false, bfr, (short)0, acc0, false, false);
            acc1 = __builtin_amdgcn_wmma_f32_16x16x32_bf16(
                       false, a1, false, bfr, (short)0, acc1, false, false);
        }
    };

    if (ASYNC) {
        // each thread owns one contiguous 32-byte chunk of the 32x64 tile:
        // rows r = tid/4, cols c0 = (tid&3)*16 .. +15   (requires a_cs == 1)
        const int r  = tid >> 2;
        const int c0 = (tid & 3) * 16;
        const __bf16* gbase = A + (size_t)(m0 + r) * a_row + a_base + c0;
        const unsigned l0 = (unsigned)(uintptr_t)(&Asm[0][0]) + (unsigned)tid * 32u;
        const unsigned l1 = (unsigned)(uintptr_t)(&Asm[1][0]) + (unsigned)tid * 32u;

        auto issue = [&](unsigned laddr, const __bf16* g) {
            // async copies 16B each; offset:16 advances BOTH LDS and global addr
            asm volatile("global_load_async_to_lds_b128 %0, %1, off"
                         :: "v"(laddr), "v"(g) : "memory");
            asm volatile("global_load_async_to_lds_b128 %0, %1, off offset:16"
                         :: "v"(laddr), "v"(g) : "memory");
        };

        issue(l0, gbase);                           // prologue: tile 0 -> buf 0
        for (int kk = 0; kk < K; kk += 64) {
            const int cur = (kk >> 6) & 1;
            __syncthreads();                        // buf[cur^1] readers done
            const bool has_next = (kk + 64) < K;
            if (has_next) issue(cur ? l0 : l1, gbase + kk + 64);
            // in-order completion: <=2 outstanding leaves only the just-issued
            // pair pending, i.e. the current tile has fully landed in LDS
            if (has_next) asm volatile("s_wait_asynccnt 0x2" ::: "memory");
            else          asm volatile("s_wait_asynccnt 0x0" ::: "memory");
            __syncthreads();                        // all waves' data visible
            mma_tile(&Asm[cur][0], kk >> 5);
        }
    } else {
        for (int kk = 0; kk < K; kk += 64) {
            #pragma unroll
            for (int j = 0; j < 16; ++j) {
                int e = tid * 16 + j;
                int rr = e >> 6, c = e & 63;
                int col = kk + c;
                int off;
                if (AMODE == 0) off = a_base + col * a_cs;
                else            off = c_grp_base[lsel][col >> 7] + (col & 127) * a_cs + a_base;
                Asm[0][e] = A[(size_t)(m0 + rr) * a_row + off];
            }
            __syncthreads();
            mma_tile(&Asm[0][0], kk >> 5);
            __syncthreads();
        }
    }

    // ---- epilogue: scale, bias, activation, residual, store ----
    const int nl = lane & 15, h = lane >> 4;
    const int ncol = nt * 16 + nl;
    const float bv = bias ? bias[ncol] : 0.f;
    #pragma unroll
    for (int mt = 0; mt < 2; ++mt) {
        v8f acc = mt ? acc1 : acc0;
        #pragma unroll
        for (int rr = 0; rr < 8; ++rr) {
            int m = m0 + mt * 16 + rr + 8 * h;
            float v = acc[rr] * scale + bv;
            if (ACT == 1) v = silu_f(v);
            size_t co = (size_t)m * c_row + c_base + (size_t)ncol * c_cs;
            if (OUTMODE == 0)      ((float*)Cp)[co]  = v + (addend ? addend[co] : 0.f);
            else if (OUTMODE == 1) ((__bf16*)Cp)[co] = (__bf16)v;
            else                   ((float*)Cp)[co]  = v;
        }
    }
}

// ---------------------------------------------------------------------------
// 'uuu' CG tensor product (one thread per node-channel) + tp-gate invariants.
// Real-basis Wigner-3j, order l=1:(y,z,x), l=2:(xy,yz,z2,xz,x2-y2);
// constants folded with the sqrt(2*lo+1) path scale (tabulated offline).
// ---------------------------------------------------------------------------
__global__ __launch_bounds__(256) void tp_uuu_kernel(
    const __bf16* __restrict__ xl, const __bf16* __restrict__ xr,
    const float* __restrict__ tpw,
    __bf16* __restrict__ tp, __bf16* __restrict__ invtp)
{
    int t = blockIdx.x * blockDim.x + threadIdx.x;
    if (t >= NODES * 128) return;
    int n = t >> 7, u = t & 127;
    const __bf16* L = xl + (size_t)n * IN_DIM;
    const __bf16* R = xr + (size_t)n * IN_DIM;

    float a0 = (float)L[u], b0 = (float)R[u];
    float a1[3], b1[3], a2[5], b2[5];
    #pragma unroll
    for (int i = 0; i < 3; ++i) { a1[i] = (float)L[128 + u*3 + i]; b1[i] = (float)R[128 + u*3 + i]; }
    #pragma unroll
    for (int i = 0; i < 5; ++i) { a2[i] = (float)L[512 + u*5 + i]; b2[i] = (float)R[512 + u*5 + i]; }

    float o[35];
    // p0 (0,0,0)
    o[0] = a0 * b0;
    // p1 (0,1,1): sqrt3 * (1/sqrt3) = 1
    o[1] = a0*b1[0]; o[2] = a0*b1[1]; o[3] = a0*b1[2];
    // p2 (0,2,2)
    #pragma unroll
    for (int i = 0; i < 5; ++i) o[4+i] = a0 * b2[i];
    // p3 (1,0,1)
    o[9] = a1[0]*b0; o[10] = a1[1]*b0; o[11] = a1[2]*b0;
    // p4 (1,1,0)
    o[12] = 0.57735027f * (a1[0]*b1[0] + a1[1]*b1[1] + a1[2]*b1[2]);
    // p5 (1,1,2): symmetric traceless vector product (x sqrt5)
    o[13] = 0.70710678f*(a1[2]*b1[0] + a1[0]*b1[2]);                         // xy
    o[14] = 0.70710678f*(a1[0]*b1[1] + a1[1]*b1[0]);                         // yz
    o[15] = 0.81649658f*a1[1]*b1[1] - 0.40824829f*(a1[0]*b1[0]+a1[2]*b1[2]); // z2
    o[16] = 0.70710678f*(a1[2]*b1[1] + a1[1]*b1[2]);                         // xz
    o[17] = 0.70710678f*(a1[2]*b1[2] - a1[0]*b1[0]);                         // x2-y2
    // p6 (1,2,1)  (permutation of p5, x sqrt3)
    o[18] = 0.54772256f*(a1[2]*b2[0] + a1[1]*b2[1]) - 0.31622777f*a1[0]*b2[2] - 0.54772256f*a1[0]*b2[4];
    o[19] = 0.54772256f*(a1[0]*b2[1] + a1[2]*b2[3]) + 0.63245553f*a1[1]*b2[2];
    o[20] = 0.54772256f*(a1[0]*b2[0] + a1[1]*b2[3]) - 0.31622777f*a1[2]*b2[2] + 0.54772256f*a1[2]*b2[4];
    // p7 (2,0,2)
    #pragma unroll
    for (int i = 0; i < 5; ++i) o[21+i] = a2[i] * b0;
    // p8 (2,1,1)  (permutation of p5, x sqrt3)
    o[26] = 0.54772256f*(a2[0]*b1[2] + a2[1]*b1[1]) - 0.31622777f*a2[2]*b1[0] - 0.54772256f*a2[4]*b1[0];
    o[27] = 0.54772256f*(a2[1]*b1[0] + a2[3]*b1[2]) + 0.63245553f*a2[2]*b1[1];
    o[28] = 0.54772256f*(a2[0]*b1[0] + a2[3]*b1[1]) - 0.31622777f*a2[2]*b1[2] + 0.54772256f*a2[4]*b1[2];
    // p9 (2,2,0)
    o[29] = 0.44721360f*(a2[0]*b2[0]+a2[1]*b2[1]+a2[2]*b2[2]+a2[3]*b2[3]+a2[4]*b2[4]);
    // p10 (2,2,2): x sqrt5 (coefficients tabulated offline)
    {
        const float T = 0.53452248f, Hh = 0.26726124f, S = 0.46291005f;
        o[30] = -T*(a2[0]*b2[2] + a2[2]*b2[0]) + S*(a2[1]*b2[3] + a2[3]*b2[1]);
        o[31] =  Hh*(a2[1]*b2[2] + a2[2]*b2[1]) + S*(a2[0]*b2[3] + a2[3]*b2[0])
               -  S*(a2[1]*b2[4] + a2[4]*b2[1]);
        o[32] =  T*a2[2]*b2[2] + Hh*(a2[1]*b2[1] + a2[3]*b2[3]) - T*(a2[0]*b2[0] + a2[4]*b2[4]);
        o[33] =  Hh*(a2[3]*b2[2] + a2[2]*b2[3]) + S*(a2[0]*b2[1] + a2[1]*b2[0])
               +  S*(a2[3]*b2[4] + a2[4]*b2[3]);
        o[34] =  Hh*(a2[4]*b2[2] + a2[2]*b2[4]) + S*(a2[1]*b2[1] - a2[3]*b2[3]);
    }

    // scale by learned path weight, emit tp values + per-path invariants
    __bf16* tpn = tp + (size_t)n * TP_DIM;
    __bf16* ivn = invtp + (size_t)n * NM_TP;
    const int obase[11] = {0,1,4,9,12,13,18,21,26,29,30};
    #pragma unroll
    for (int p = 0; p < 11; ++p) {
        float tw = tpw[p * 128 + u];
        int w = c_tp_w[p], ob = obase[p], base = c_tp_base[p];
        float s = 0.f, first = 0.f;
        #pragma unroll
        for (int k = 0; k < 5; ++k) {
            if (k < w) {
                float v = o[ob + k] * tw;
                if (k == 0) first = v;
                tpn[base + u * w + k] = (__bf16)v;
                s += v * v;
            }
        }
        ivn[p * 128 + u] = (__bf16)((w == 1) ? first : sqrtf(s));
    }
}

// ---------------------------------------------------------------------------
// Gate the tp output with g_p -> bf16 (input to linear_p GEMMs)
// ---------------------------------------------------------------------------
__global__ void apply_gate_p(const __bf16* __restrict__ tp, const float* __restrict__ gp,
                             __bf16* __restrict__ tpg) {
    int t = blockIdx.x * blockDim.x + threadIdx.x;
    if (t >= NODES * TP_DIM) return;
    int n = t / TP_DIM, col = t % TP_DIM;
    int p = 0;
    while (col >= c_tp_base[p + 1]) ++p;
    int u = (col - c_tp_base[p]) / c_tp_w[p];
    tpg[t] = (__bf16)((float)tp[t] * gp[(size_t)n * NM_TP + p * 128 + u]);
}

// ---------------------------------------------------------------------------
// Host-side orchestration
// ---------------------------------------------------------------------------
extern "C" void kernel_launch(void* const* d_in, const int* in_sizes, int n_in,
                              void* d_out, int out_size, void* d_ws, size_t ws_size,
                              hipStream_t stream)
{
    (void)in_sizes; (void)n_in; (void)out_size; (void)ws_size;

    const float* x    = (const float*)d_in[0];
    const float* fii  = (const float*)d_in[1];
    const float* Wg_l = (const float*)d_in[2];
    const float* bg_l = (const float*)d_in[3];
    const float* Wl[6] = { (const float*)d_in[4],  (const float*)d_in[5],  (const float*)d_in[6],
                           (const float*)d_in[10], (const float*)d_in[11], (const float*)d_in[12] };
    const float* bl0  = (const float*)d_in[7];
    const float* Wg_r = (const float*)d_in[8];
    const float* bg_r = (const float*)d_in[9];
    const float* br0  = (const float*)d_in[13];
    const float* tpw  = (const float*)d_in[14];
    const float* Wg_p = (const float*)d_in[15];
    const float* bg_p = (const float*)d_in[16];
    const float* Wp[3] = { (const float*)d_in[17], (const float*)d_in[18], (const float*)d_in[19] };

    char* ws = (char*)d_ws;
    size_t off = 0;
    auto alloc_bf = [&](size_t n) -> __bf16* {
        off = (off + 255) & ~(size_t)255; __bf16* p = (__bf16*)(ws + off); off += n * 2; return p; };
    auto alloc_f = [&](size_t n) -> float* {
        off = (off + 255) & ~(size_t)255; float* p = (float*)(ws + off); off += n * 4; return p; };

    // workspace layout
    __bf16* wgl_bf  = alloc_bf(NM_IN * NM_IN);
    __bf16* wgr_bf  = alloc_bf(NM_IN * NM_IN);
    __bf16* wl_bf   = alloc_bf(6 * 128 * 128);
    __bf16* wgp_bf  = alloc_bf((size_t)NM_TP * NM_TP);
    __bf16* wp_bf[3] = { alloc_bf(384 * 128), alloc_bf(512 * 128), alloc_bf(512 * 128) };
    __bf16* inv_bf  = alloc_bf((size_t)NODES * NM_IN);
    float*  gl_f    = alloc_f ((size_t)NODES * NM_IN);
    float*  gr_f    = alloc_f ((size_t)NODES * NM_IN);
    __bf16* xgl_bf  = alloc_bf((size_t)NODES * IN_DIM);
    __bf16* xgr_bf  = alloc_bf((size_t)NODES * IN_DIM);
    __bf16* xl_bf   = alloc_bf((size_t)NODES * IN_DIM);
    __bf16* xr_bf   = alloc_bf((size_t)NODES * IN_DIM);
    __bf16* tp_bf   = alloc_bf((size_t)NODES * TP_DIM);
    __bf16* invtp_bf= alloc_bf((size_t)NODES * NM_TP);
    float*  gp_f    = alloc_f ((size_t)NODES * NM_TP);
    __bf16* tpg_bf  = alloc_bf((size_t)NODES * TP_DIM);

    auto pack = [&](const float* s, __bf16* d, int K, int N) {
        pack_b<<<(K * N + 255) / 256, 256, 0, stream>>>(s, d, K, N); };

    // 0) weights -> packed bf16 WMMA B-fragment layout
    pack(Wg_l, wgl_bf, NM_IN, NM_IN);
    pack(Wg_r, wgr_bf, NM_IN, NM_IN);
    for (int i = 0; i < 6; ++i) pack(Wl[i], wl_bf + i * 16384, 128, 128);
    pack(Wg_p, wgp_bf, NM_TP, NM_TP);
    pack(Wp[0], wp_bf[0], 384, 128);
    pack(Wp[1], wp_bf[1], 512, 128);
    pack(Wp[2], wp_bf[2], 512, 128);

    // 1) invariants of x
    invariants_x<<<(NODES * NM_IN + 255) / 256, 256, 0, stream>>>(x, inv_bf);

    // 2) gate GEMMs: g = silu(inv @ Wg + bg)   [16384x384]x[384x384]  (async A)
    gemm_wmma<0, 2, 1, 1><<<dim3(NM_IN / 64, NODES / 32), 128, 0, stream>>>(
        inv_bf, NM_IN, 0, 1, wgl_bf, NM_IN, bg_l, 1.f, gl_f, NM_IN, 0, 1, nullptr, NM_IN, 0);
    gemm_wmma<0, 2, 1, 1><<<dim3(NM_IN / 64, NODES / 32), 128, 0, stream>>>(
        inv_bf, NM_IN, 0, 1, wgr_bf, NM_IN, bg_r, 1.f, gr_f, NM_IN, 0, 1, nullptr, NM_IN, 0);

    // 3) apply gates
    apply_gate_lr<<<(NODES * IN_DIM + 255) / 256, 256, 0, stream>>>(x, gl_f, gr_f, xgl_bf, xgr_bf);

    // 4) linear_sq: 18 x [16384x128]x[128x128] over strided irrep views
    const int offl[3] = {0, 128, 512};
    for (int side = 0; side < 2; ++side) {
        const __bf16* Ain = side ? xgr_bf : xgl_bf;
        __bf16* Cout      = side ? xr_bf : xl_bf;
        // l = 0 : contiguous columns -> async A staging
        gemm_wmma<0, 1, 0, 1><<<dim3(2, NODES / 32), 128, 0, stream>>>(
            Ain, IN_DIM, 0, 1, wl_bf + (side * 3) * 16384, 128,
            side ? br0 : bl0, 0.08838834764f,
            Cout, IN_DIM, 0, 1, nullptr, 128, 0);
        // l = 1,2 : strided irrep views -> sync gather staging
        for (int l = 1; l < 3; ++l)
            for (int i = 0; i < 2 * l + 1; ++i)
                gemm_wmma<0, 1, 0, 0><<<dim3(2, NODES / 32), 128, 0, stream>>>(
                    Ain, IN_DIM, offl[l] + i, 2 * l + 1, wl_bf + (side * 3 + l) * 16384, 128,
                    nullptr, 0.08838834764f,
                    Cout, IN_DIM, offl[l] + i, 2 * l + 1, nullptr, 128, 0);
    }

    // 5) CG tensor product + tp-gate invariants
    tp_uuu_kernel<<<(NODES * 128 + 255) / 256, 256, 0, stream>>>(xl_bf, xr_bf, tpw, tp_bf, invtp_bf);

    // 6) big gate GEMM: [16384x1408]x[1408x1408] (65-GFLOP hot spot, async A)
    gemm_wmma<0, 2, 1, 1><<<dim3(NM_TP / 64, NODES / 32), 128, 0, stream>>>(
        invtp_bf, NM_TP, 0, 1, wgp_bf, NM_TP, bg_p, 1.f, gp_f, NM_TP, 0, 1, nullptr, NM_TP, 0);

    // 7) gate the tp output
    apply_gate_p<<<(NODES * TP_DIM + 255) / 256, 256, 0, stream>>>(tp_bf, gp_f, tpg_bf);

    // 8) linear_p + fii_in residual: 9 GEMMs over grouped tp columns
    const float pscale[3] = {0.051031036307983f, 0.044194173824159f, 0.044194173824159f};
    const int   Kl[3]     = {384, 512, 512};
    for (int l = 0; l < 3; ++l)
        for (int i = 0; i < 2 * l + 1; ++i)
            gemm_wmma<1, 0, 0, 0><<<dim3(2, NODES / 32), 128, 0, stream>>>(
                tpg_bf, TP_DIM, i, 2 * l + 1, wp_bf[l], 128, nullptr, pscale[l],
                (float*)d_out, IN_DIM, offl[l] + i, 2 * l + 1, fii, Kl[l], l);
}
